// NumToLang_LSTM_35055523070751
// MI455X (gfx1250) — matrix operand order repaired and checked by reading, not compiled, not run on hardware
//
#include <hip/hip_runtime.h>
#include <hip/hip_bf16.h>

// Problem constants (reference: B=1024, T=64, H=512, V=1024, GEN=100)
#define B_   1024
#define T_   64
#define H_   512
#define V_   1024
#define G4H  2048   // 4*H
#define GEN_ 100
#define GENP 128    // GEN padded to multiple of 32 (K) / 64 (N-strip)

typedef __attribute__((ext_vector_type(16))) __bf16 bf16x16;
typedef __attribute__((ext_vector_type(8)))  __bf16 bf16x8;
typedef __attribute__((ext_vector_type(8)))  float  f32x8;

__device__ __forceinline__ float sigmoidf_(float x) {
    return 1.0f / (1.0f + __expf(-x));
}

// ---------------------------------------------------------------------------
// Core WMMA tile loop: one wave computes 4 16x16 output tiles (N-tiles
// ntile0 + j*nstride) sharing one A fragment per K-step.
// A row-major bf16 [M,K]; Bp packed: ((ntile*ksteps + ks)*32 + lane)*16 + e.
// ISA 7.12.2 (16-bit A 16x32): lanes 0-15 hold K {0..7,16..23},
// lanes 16-31 hold K {8..15,24..31} -> two contiguous b128 loads per lane.
// ---------------------------------------------------------------------------
__device__ __forceinline__ void wmma_acc4(const __bf16* __restrict__ A,
                                          const __bf16* __restrict__ Bp,
                                          int K, int mtile, int ntile0,
                                          int nstride, f32x8 acc[4])
{
    const int lane   = threadIdx.x & 31;
    const int mrow   = mtile * 16 + (lane & 15);
    const int khalf  = (lane & 16) ? 8 : 0;
    const int ksteps = K >> 5;
    const __bf16* arow = A + (size_t)mrow * K;

    for (int ks = 0; ks < ksteps; ++ks) {
        const int k0 = ks << 5;
        bf16x8 alo = *(const bf16x8*)(arow + k0 + khalf);
        bf16x8 ahi = *(const bf16x8*)(arow + k0 + 16 + khalf);
        bf16x16 a;
#pragma unroll
        for (int i = 0; i < 8; ++i) { a[i] = alo[i]; a[i + 8] = ahi[i]; }
#pragma unroll
        for (int j = 0; j < 4; ++j) {
            const __bf16* bptr =
                Bp + (((size_t)(ntile0 + j * nstride) * ksteps + ks) * 32 + lane) * 16;
            if (ks + 1 < ksteps)
                __builtin_prefetch(bptr + 32 * 16, 0, 3);  // next K-step, WGP scope
            bf16x16 b = *(const bf16x16*)bptr;
            acc[j] = __builtin_amdgcn_wmma_f32_16x16x32_bf16(
                false, a, false, b, (short)0, acc[j], false, false);
        }
    }
}

// ---------------------------------------------------------------------------
// Weight packing: fp32 [nvalid x kvalid] row-major (leading dim srcld) ->
// bf16 WMMA B-operand layout, zero-padded.  ISA 7.12.2: B 32x16, lanes 0-15
// hold K {0..7,16..23}, lanes 16-31 hold K {8..15,24..31}, N = lane%16.
// ---------------------------------------------------------------------------
__global__ void k_pack_b(const float* __restrict__ src, __bf16* __restrict__ dst,
                         int nvalid, int kvalid, int srcld, int ksteps, int total)
{
    int idx = blockIdx.x * blockDim.x + threadIdx.x;
    if (idx >= total) return;
    int e     = idx & 15;
    int lane  = (idx >> 4) & 31;
    int ks    = (idx >> 9) % ksteps;
    int ntile = idx / (512 * ksteps);
    int n = ntile * 16 + (lane & 15);
    int k = ks * 32 + ((lane & 16) ? 8 : 0) + (e < 8 ? e : e + 8);
    float v = (n < nvalid && k < kvalid) ? src[(size_t)n * srcld + k] : 0.0f;
    dst[idx] = (__bf16)v;
}

__global__ void k_bias(const float* __restrict__ b_ih,
                       const float* __restrict__ b_hh,
                       float* __restrict__ bias)
{
    int i = blockIdx.x * blockDim.x + threadIdx.x;
    if (i < G4H) bias[i] = b_ih[i] + b_hh[i];
}

// h0 = input*Wh + bh (bf16 for GEMM A), c0 = input*Wc + bc (fp32 state)
__global__ void k_init(const float* __restrict__ input,
                       const float* __restrict__ Wh, const float* __restrict__ bh,
                       const float* __restrict__ Wc, const float* __restrict__ bc,
                       __bf16* __restrict__ h, float* __restrict__ c)
{
    int idx = blockIdx.x * blockDim.x + threadIdx.x;
    if (idx >= B_ * H_) return;
    int b = idx / H_, j = idx % H_;
    float x = input[b];
    h[idx] = (__bf16)(x * Wh[j] + bh[j]);
    c[idx] = x * Wc[j] + bc[j];
}

// tok[b] = argmax of one-hot row onehots[b, 0, :]   (one wave32 per row)
__global__ void k_init_tok(const float* __restrict__ onehots, int* __restrict__ tok)
{
    int row = (blockIdx.x * blockDim.x + threadIdx.x) >> 5;
    if (row >= B_) return;
    int lane = threadIdx.x & 31;
    const float* r = onehots + (size_t)row * T_ * V_;
    int best = 0x7fffffff;
    for (int v = lane; v < V_; v += 32)
        if (r[v] > 0.5f && v < best) best = v;
#pragma unroll
    for (int off = 16; off > 0; off >>= 1)
        best = min(best, __shfl_xor(best, off, 32));
    if (lane == 0) tok[row] = (best == 0x7fffffff) ? 0 : best;
}

// ---------------------------------------------------------------------------
// Fused LSTM cell: each wave's 4 accumulators are the SAME j-tile of gates
// i,f,g,o (N-tiles jt, 32+jt, 64+jt, 96+jt -> N-stride 32).  Epilogue fuses
// the one-hot W_ih gather, bias, and the full gate nonlinearity, updating
// c (fp32, in place, race-free) and writing h' into the ping-pong buffer.
// Eliminates the 8 MB/step g tensor round-trip and one kernel per step.
// ---------------------------------------------------------------------------
__global__ void k_cell_fused(const __bf16* __restrict__ hin,
                             const __bf16* __restrict__ Wp,
                             const float* __restrict__ bias,
                             const float* __restrict__ W_ih,
                             const int* __restrict__ tok,
                             float* __restrict__ c,
                             __bf16* __restrict__ hout)
{
    const int wave  = (blockIdx.x * blockDim.x + threadIdx.x) >> 5;
    const int mtile = wave >> 5;          // 64 m-tiles
    const int jt    = wave & 31;          // 32 j-tiles (H/16)
    f32x8 acc[4] = {};
    wmma_acc4(hin, Wp, H_, mtile, jt, H_ / 16, acc);   // gate g at ntile g*32+jt

    const int lane  = threadIdx.x & 31;
    const int jcol  = jt * 16 + (lane & 15);
    const int mbase = mtile * 16 + ((lane & 16) ? 8 : 0);

    int t8[8];
#pragma unroll
    for (int r = 0; r < 8; ++r) t8[r] = tok[mbase + r];

    float bn[4];
    const float* wr[4];
#pragma unroll
    for (int gte = 0; gte < 4; ++gte) {
        const int n = gte * H_ + jcol;
        bn[gte] = bias[n];
        wr[gte] = W_ih + (size_t)n * V_;   // W_ih row n, gathered by token
    }

#pragma unroll
    for (int r = 0; r < 8; ++r) {
        const int t = t8[r];
        float gi = acc[0][r] + bn[0] + wr[0][t];
        float gf = acc[1][r] + bn[1] + wr[1][t];
        float gg = acc[2][r] + bn[2] + wr[2][t];
        float go = acc[3][r] + bn[3] + wr[3][t];
        const size_t idx = (size_t)(mbase + r) * H_ + jcol;
        float cn = sigmoidf_(gf) * c[idx] + sigmoidf_(gi) * tanhf(gg);
        c[idx] = cn;
        hout[idx] = (__bf16)(sigmoidf_(go) * tanhf(cn));
    }
}

// z1 = relu(h @ W1^T + b1), N padded 100->128 with zeros (so head2 K=128 works)
__global__ void k_head1(const __bf16* __restrict__ h, const __bf16* __restrict__ W1p,
                        const float* __restrict__ b1, __bf16* __restrict__ z1)
{
    const int wave   = (blockIdx.x * blockDim.x + threadIdx.x) >> 5;
    const int NG     = GENP / 64;            // 2
    const int mtile  = wave / NG;
    const int ntile0 = (wave % NG) * 4;
    f32x8 acc[4] = {};
    wmma_acc4(h, W1p, H_, mtile, ntile0, 1, acc);

    const int lane  = threadIdx.x & 31;
    const int mbase = mtile * 16 + ((lane & 16) ? 8 : 0);
#pragma unroll
    for (int j = 0; j < 4; ++j) {
        const int   n  = (ntile0 + j) * 16 + (lane & 15);
        const float bn = (n < GEN_) ? b1[n] : 0.0f;
#pragma unroll
        for (int r = 0; r < 8; ++r) {
            float v = (n < GEN_) ? fmaxf(acc[j][r] + bn, 0.0f) : 0.0f;
            z1[(size_t)(mbase + r) * GENP + n] = (__bf16)v;
        }
    }
}

// ---------------------------------------------------------------------------
// Fused head2 + log-softmax + argmax: one 16-wave block owns a full
// 16x1024 logits strip (wave w = N-tiles 4w..4w+3, K=128).  Row statistics
// reduced half-wave via __shfl_xor then cross-wave via LDS; first-occurrence
// argmax via LDS atomicMin.  Eliminates the 4 MB/step logits round-trip and
// two kernels per step; tok for the next step written here.
// ---------------------------------------------------------------------------
__global__ __launch_bounds__(512) void k_head2_softmax(
    const __bf16* __restrict__ z1, const __bf16* __restrict__ W2p,
    const float* __restrict__ b2, float* __restrict__ out,
    int* __restrict__ tok)
{
    const int mtile  = blockIdx.x;        // 64 blocks, 16 rows each
    const int wave   = threadIdx.x >> 5;  // 16 waves -> full V=1024 row
    const int lane   = threadIdx.x & 31;
    const int ntile0 = wave * 4;
    f32x8 acc[4] = {};
    wmma_acc4(z1, W2p, GENP, mtile, ntile0, 1, acc);

    __shared__ float red[16][16];   // [wave][row]
    __shared__ float rowmax[16];
    __shared__ float lse[16];
    __shared__ int   smin[16];

    const int half = (lane & 16) ? 8 : 0;

#pragma unroll
    for (int j = 0; j < 4; ++j) {
        const float bn = b2[(ntile0 + j) * 16 + (lane & 15)];
#pragma unroll
        for (int r = 0; r < 8; ++r) acc[j][r] += bn;
    }

    // per-row max over this wave's 64 columns (independent 16-lane halves)
#pragma unroll
    for (int r = 0; r < 8; ++r) {
        float lm = fmaxf(fmaxf(acc[0][r], acc[1][r]), fmaxf(acc[2][r], acc[3][r]));
#pragma unroll
        for (int off = 8; off > 0; off >>= 1) lm = fmaxf(lm, __shfl_xor(lm, off, 32));
        if ((lane & 15) == 0) red[wave][half + r] = lm;
    }
    __syncthreads();
    if (threadIdx.x < 16) {
        float m = red[0][threadIdx.x];
        for (int w = 1; w < 16; ++w) m = fmaxf(m, red[w][threadIdx.x]);
        rowmax[threadIdx.x] = m;
        smin[threadIdx.x] = 0x7fffffff;
    }
    __syncthreads();

    // per-row sum of exp
#pragma unroll
    for (int r = 0; r < 8; ++r) {
        const float rm = rowmax[half + r];
        float s = __expf(acc[0][r] - rm) + __expf(acc[1][r] - rm) +
                  __expf(acc[2][r] - rm) + __expf(acc[3][r] - rm);
#pragma unroll
        for (int off = 8; off > 0; off >>= 1) s += __shfl_xor(s, off, 32);
        if ((lane & 15) == 0) red[wave][half + r] = s;
    }
    __syncthreads();
    if (threadIdx.x < 16) {
        float s = 0.0f;
        for (int w = 0; w < 16; ++w) s += red[w][threadIdx.x];
        lse[threadIdx.x] = rowmax[threadIdx.x] + __logf(s);
    }
    __syncthreads();

    // write log-probs; first-occurrence argmax -> tok for next step
#pragma unroll
    for (int j = 0; j < 4; ++j) {
        const int n = (ntile0 + j) * 16 + (lane & 15);
#pragma unroll
        for (int r = 0; r < 8; ++r) {
            const int row = half + r;
            const float val = acc[j][r];
            out[(size_t)(mtile * 16 + row) * V_ + n] = val - lse[row];
            if (val == rowmax[row]) atomicMin(&smin[row], n);
        }
    }
    __syncthreads();
    if (threadIdx.x < 16) tok[mtile * 16 + threadIdx.x] = smin[threadIdx.x];
}

// ---------------------------------------------------------------------------
extern "C" void kernel_launch(void* const* d_in, const int* in_sizes, int n_in,
                              void* d_out, int out_size, void* d_ws, size_t ws_size,
                              hipStream_t stream)
{
    (void)in_sizes; (void)n_in; (void)out_size; (void)ws_size;
    const float* input   = (const float*)d_in[0];
    const float* onehots = (const float*)d_in[1];
    // d_in[2] digits, d_in[3] teacher (==0): unused (greedy path)
    const float* Wh   = (const float*)d_in[4];
    const float* bh   = (const float*)d_in[5];
    const float* Wc   = (const float*)d_in[6];
    const float* bc   = (const float*)d_in[7];
    const float* W_ih = (const float*)d_in[8];
    const float* W_hh = (const float*)d_in[9];
    const float* b_ih = (const float*)d_in[10];
    const float* b_hh = (const float*)d_in[11];
    const float* W1   = (const float*)d_in[12];
    const float* b1   = (const float*)d_in[13];
    const float* W2   = (const float*)d_in[14];
    const float* b2   = (const float*)d_in[15];
    float* out = (float*)d_out;

    // workspace carve-up (256B aligned slices)
    char* wsb = (char*)d_ws;
    size_t off = 0;
    auto carve = [&](size_t bytes) -> void* {
        void* p = wsb + off;
        off = (off + bytes + 255) & ~(size_t)255;
        return p;
    };
    __bf16* Whhp = (__bf16*)carve((size_t)G4H * H_ * 2);   // packed W_hh
    __bf16* W1p  = (__bf16*)carve((size_t)GENP * H_ * 2);  // packed W1 (N-pad)
    __bf16* W2p  = (__bf16*)carve((size_t)V_ * GENP * 2);  // packed W2 (K-pad)
    float*  bias = (float*) carve((size_t)G4H * 4);
    __bf16* hA   = (__bf16*)carve((size_t)B_ * H_ * 2);    // h ping-pong
    __bf16* hB   = (__bf16*)carve((size_t)B_ * H_ * 2);
    float*  cst  = (float*) carve((size_t)B_ * H_ * 4);
    __bf16* z1   = (__bf16*)carve((size_t)B_ * GENP * 2);
    int*    tok  = (int*)   carve((size_t)B_ * 4);

    const int TB = 256;
    // one-time weight packing + state init
    k_pack_b<<<(G4H * H_ + TB - 1) / TB, TB, 0, stream>>>(
        W_hh, Whhp, G4H, H_, H_, H_ / 32, G4H * H_);
    k_pack_b<<<(GENP * H_ + TB - 1) / TB, TB, 0, stream>>>(
        W1, W1p, GEN_, H_, H_, H_ / 32, GENP * H_);
    k_pack_b<<<(V_ * GENP + TB - 1) / TB, TB, 0, stream>>>(
        W2, W2p, V_, GEN_, GEN_, GENP / 32, V_ * GENP);
    k_bias<<<(G4H + TB - 1) / TB, TB, 0, stream>>>(b_ih, b_hh, bias);
    k_init<<<(B_ * H_ + TB - 1) / TB, TB, 0, stream>>>(input, Wh, bh, Wc, bc, hA, cst);
    k_init_tok<<<B_ / 8, TB, 0, stream>>>(onehots, tok);

    // 64 serial decode steps, 3 kernels each (stream order = dependency chain)
    __bf16* hin = hA;
    __bf16* hout = hB;
    for (int t = 0; t < T_; ++t) {
        k_cell_fused<<<(64 * 32 * 32) / TB, TB, 0, stream>>>(
            hin, Whhp, bias, W_ih, tok, cst, hout);               // 256 blocks
        k_head1<<<(64 * (GENP / 64) * 32) / TB, TB, 0, stream>>>(
            hout, W1p, b1, z1);                                   // 16 blocks
        k_head2_softmax<<<64, 512, 0, stream>>>(
            z1, W2p, b2, out + (size_t)t * B_ * V_, tok);         // 64 blocks
        __bf16* tmp = hin; hin = hout; hout = tmp;
    }
}
